// GatNet1_27307402068442
// MI455X (gfx1250) — matrix operand order repaired and checked by reading, not compiled, not run on hardware
//
#include <hip/hip_runtime.h>
#include <cstring>
#include <cstdint>

// ---------------------------------------------------------------------------
// GAT + SAGPool + GAT + global-attention readout for MI455X (gfx1250).
// Dense projections use v_wmma_f32_16x16x32_bf16 (fp32 accumulate).
// A tiles staged via LDS in fragment order (8-way wave reuse, ds_load_b128);
// B fragments built directly from global (no reuse within a block).
// ---------------------------------------------------------------------------

#define NEGV (-1e9f)

typedef __bf16 bf16x16 __attribute__((ext_vector_type(16)));
typedef float  f32x8   __attribute__((ext_vector_type(8)));

__device__ __forceinline__ unsigned short f2bf(float f) {
    unsigned u = __builtin_bit_cast(unsigned, f);
    unsigned r = u + 0x7FFFu + ((u >> 16) & 1u);   // round-to-nearest-even
    return (unsigned short)(r >> 16);
}
// monotonic float -> uint key (order preserving), and inverse
__device__ __forceinline__ unsigned fkey(float f) {
    unsigned u = __builtin_bit_cast(unsigned, f);
    return u ^ ((u >> 31) ? 0xFFFFFFFFu : 0x80000000u);
}
__device__ __forceinline__ float keyf(unsigned k) {
    unsigned u = (k >> 31) ? (k ^ 0x80000000u) : ~k;
    return __builtin_bit_cast(float, u);
}
__device__ __forceinline__ float lrelu(float x) { return x > 0.f ? x : 0.4f * x; }

// ---------------------------------------------------------------------------
// C[M,128] = A[M,Kd](f32,row-major) x B[Kd,128](f32,row-major), bf16 WMMA.
// Block = 256 threads = 8 waves. Block owns a 32-row strip (two 16-row
// tiles); wave w owns columns [16w,16w+16) and issues 2 WMMAs per K-step,
// sharing one B fragment. A is staged via LDS in fragment order:
//   A element (row r, k):  lane = (r&15) + 16*((k>>3)&1),  slot = (k&7)+8*(k>>4)
// so each lane reads its bf16x16 A fragment as one contiguous 32B LDS load.
// B fragment (k, col c):   lane = (c&15) + 16*((k>>3)&1),  slot = (k&7)+8*(k>>4)
// is loaded straight from global into registers (tile covered exactly once).
// ---------------------------------------------------------------------------
__global__ __launch_bounds__(256) void gemm_wmma_n128(
    const float* __restrict__ A, const float* __restrict__ B,
    float* __restrict__ C, int M, int Kd) {
    __shared__ __attribute__((aligned(32))) unsigned short Af[2 * 512]; // [rowtile][lane][slot]
    const int tid  = threadIdx.x;
    const int wave = tid >> 5;
    const int lane = tid & 31;
    const int r0   = blockIdx.x * 32;
    const int colw = (wave << 4) + (lane & 15);   // this lane's B/C column
    const int kb   = (lane >> 4) << 3;            // B fragment K base: 0 or 8
    // A staging coords (fixed per thread)
    const int sr   = (tid * 4) >> 5;              // staged row 0..31
    const int skk  = (tid * 4) & 31;              // staged K (multiple of 4)
    const int sofs = ((sr >> 4) * 512) + (((sr & 15) + (((skk >> 3) & 1) << 4)) * 16)
                   + ((skk & 7) + ((skk >> 4) << 3));
    f32x8 acc0 = {}, acc1 = {};
    for (int k0 = 0; k0 < Kd; k0 += 32) {
        // ---- stage A: one float4 (4 consecutive K) per thread, packed b64 store
        {
            int gr = r0 + sr;
            float4 v = make_float4(0.f, 0.f, 0.f, 0.f);
            if (gr < M) v = *(const float4*)(A + (size_t)gr * Kd + k0 + skk);
            unsigned lo = (unsigned)f2bf(v.x) | ((unsigned)f2bf(v.y) << 16);
            unsigned hi = (unsigned)f2bf(v.z) | ((unsigned)f2bf(v.w) << 16);
            *(uint2*)&Af[sofs] = make_uint2(lo, hi);
        }
        // ---- B fragment straight from global (16 rows of the weight tile)
        bf16x16 b;
        #pragma unroll
        for (int i = 0; i < 16; ++i)
            b[i] = __builtin_bit_cast(__bf16, f2bf(B[(size_t)(k0 + kb + (i & 7) + ((i >> 3) << 4)) * 128 + colw]));
        __syncthreads();
        bf16x16 a0 = ((const bf16x16*)(Af +   0))[lane];
        bf16x16 a1 = ((const bf16x16*)(Af + 512))[lane];
        acc0 = __builtin_amdgcn_wmma_f32_16x16x32_bf16(
            false, a0, false, b, (short)0, acc0, false, false);
        acc1 = __builtin_amdgcn_wmma_f32_16x16x32_bf16(
            false, a1, false, b, (short)0, acc1, false, false);
        __syncthreads();
    }
    // C layout: lane%16 = N, vgpr = M (+8 for hi lane half)
    float* Cp = C + (size_t)r0 * 128 + colw;
    if (r0 + 32 <= M) {                           // full-tile fast path
        #pragma unroll
        for (int v = 0; v < 8; ++v) {
            int m = (lane < 16) ? v : v + 8;
            Cp[(size_t)m * 128]        = acc0[v];
            Cp[(size_t)(m + 16) * 128] = acc1[v];
        }
    } else {
        #pragma unroll
        for (int v = 0; v < 8; ++v) {
            int m = (lane < 16) ? v : v + 8;
            if (r0 + m      < M) Cp[(size_t)m * 128]        = acc0[v];
            if (r0 + 16 + m < M) Cp[(size_t)(m + 16) * 128] = acc1[v];
        }
    }
}

// ---------------------------------------------------------------------------
__global__ void fill_u32(unsigned* p, unsigned v, size_t n) {
    size_t i = (size_t)blockIdx.x * blockDim.x + threadIdx.x;
    if (i < n) p[i] = v;
}
__global__ void init_bias(float* out, const float* __restrict__ b, size_t nodes) {
    size_t i = (size_t)blockIdx.x * blockDim.x + threadIdx.x;
    if (i < nodes * 128) out[i] = b[i & 127];
}
// per-(node,head) attention projections el/er
__global__ void attn_proj(const float* __restrict__ hfeat, const float* __restrict__ al,
                          const float* __restrict__ ar, float* el, float* er, int n) {
    int idx = blockIdx.x * blockDim.x + threadIdx.x;
    if (idx >= n * 4) return;
    int node = idx >> 2, h = idx & 3;
    const float* hp = hfeat + (size_t)node * 128 + h * 32;
    float sl = 0.f, sr = 0.f;
    #pragma unroll
    for (int d = 0; d < 32; ++d) { float v = hp[d]; sl += v * al[h * 32 + d]; sr += v * ar[h * 32 + d]; }
    el[idx] = sl; er[idx] = sr;
}
// pass 1: segment max of leaky-relu edge logits (uint-key atomicMax)
__global__ void edge_logit_max(const int* __restrict__ src, const int* __restrict__ dst,
                               const int* __restrict__ valid, const float* __restrict__ el,
                               const float* __restrict__ er, unsigned* mkey, int E) {
    int idx = blockIdx.x * blockDim.x + threadIdx.x;
    if (idx >= E * 4) return;
    int e = idx >> 2, h = idx & 3;
    float v;
    if (valid && !valid[e]) v = NEGV;
    else v = lrelu(el[src[e] * 4 + h] + er[dst[e] * 4 + h]);
    atomicMax(&mkey[dst[e] * 4 + h], fkey(v));
}
// pass 2: ex = exp(e - m)*valid ; segment sum
__global__ void edge_exp_sum(const int* __restrict__ src, const int* __restrict__ dst,
                             const int* __restrict__ valid, const float* __restrict__ el,
                             const float* __restrict__ er, const unsigned* __restrict__ mkey,
                             float* ex, float* ssum, int E) {
    int idx = blockIdx.x * blockDim.x + threadIdx.x;
    if (idx >= E * 4) return;
    int e = idx >> 2, h = idx & 3;
    float x = 0.f;
    if (!valid || valid[e]) {
        float t = lrelu(el[src[e] * 4 + h] + er[dst[e] * 4 + h]);
        float m = keyf(mkey[dst[e] * 4 + h]);
        x = __expf(t - m);
        atomicAdd(&ssum[dst[e] * 4 + h], x);
    }
    ex[e * 4 + h] = x;
}
// pass 3: out[dst] += alpha * h[src]; optionally emit alpha (attention output)
__global__ __launch_bounds__(256) void edge_aggregate(
    const int* __restrict__ src, const int* __restrict__ dst,
    const float* __restrict__ ex, const float* __restrict__ ssum,
    const float* __restrict__ hfeat, float* out, float* atten, int E) {
    int t = threadIdx.x & 127;
    int e = blockIdx.x * 2 + (threadIdx.x >> 7);
    if (e >= E) return;
    int h = t >> 5, d = t & 31;
    int s = src[e], dn = dst[e];
    float a = ex[e * 4 + h] / fmaxf(ssum[dn * 4 + h], 1e-16f);
    if (atten && d == 0) atten[e * 4 + h] = a;
    if (a != 0.f) atomicAdd(&out[(size_t)dn * 128 + t], a * hfeat[(size_t)s * 128 + t]);
}
// ---------------------------------------------------------------------------
__global__ void degrees(const int* __restrict__ src, const int* __restrict__ dst,
                        float* dego, float* degi, int E) {
    int e = blockIdx.x * blockDim.x + threadIdx.x;
    if (e >= E) return;
    atomicAdd(&dego[src[e]], 1.f);
    atomicAdd(&degi[dst[e]], 1.f);
}
// hc[n] = (x[n]·Wc) * deg_o^-1/2   (one wave per node)
__global__ __launch_bounds__(256) void conv_score_a(const float* __restrict__ x,
                                                    const float* __restrict__ Wc,
                                                    const float* __restrict__ dego,
                                                    float* hc, int n) {
    int wv = (blockIdx.x * 256 + threadIdx.x) >> 5;
    int lane = threadIdx.x & 31;
    if (wv >= n) return;
    float s = 0.f;
    #pragma unroll
    for (int j = 0; j < 4; ++j) { int d = lane + 32 * j; s += x[(size_t)wv * 128 + d] * Wc[d]; }
    #pragma unroll
    for (int m = 16; m >= 1; m >>= 1) s += __shfl_xor(s, m, 32);
    if (lane == 0) hc[wv] = s * rsqrtf(fmaxf(dego[wv], 1.f));
}
__global__ void edge_agg_scalar(const int* __restrict__ src, const int* __restrict__ dst,
                                const float* __restrict__ hc, float* agg, int E) {
    int e = blockIdx.x * blockDim.x + threadIdx.x;
    if (e >= E) return;
    atomicAdd(&agg[dst[e]], hc[src[e]]);
}
__global__ void score_keys(const float* __restrict__ agg, const float* __restrict__ degi,
                           const float* __restrict__ bc, float* score, unsigned* keys, int n) {
    int i = blockIdx.x * blockDim.x + threadIdx.x;
    if (i >= n) return;
    float s = agg[i] * rsqrtf(fmaxf(degi[i], 1.f)) + bc[0];
    score[i] = s; keys[i] = fkey(s);
}
// ---- exact deterministic top-K (2x16-bit radix histogram select) ----------
__global__ void hist_hi(const unsigned* __restrict__ keys, unsigned* hist, int n) {
    int i = blockIdx.x * blockDim.x + threadIdx.x;
    if (i < n) atomicAdd(&hist[keys[i] >> 16], 1u);
}
__global__ void hist_lo(const unsigned* __restrict__ keys, const unsigned* __restrict__ scal,
                        unsigned* hist, int n) {
    int i = blockIdx.x * blockDim.x + threadIdx.x;
    if (i >= n) return;
    unsigned k = keys[i];
    if ((k >> 16) == scal[0]) atomicAdd(&hist[k & 0xFFFFu], 1u);
}
__global__ void scan_hist_hi(const unsigned* __restrict__ hist, unsigned* scal, int Ksel) {
    unsigned cum = 0;
    for (int b = 65535; b >= 0; --b) {
        unsigned c = hist[b];
        if (cum + c >= (unsigned)Ksel) { scal[0] = (unsigned)b; scal[1] = cum; return; }
        cum += c;
    }
    scal[0] = 0; scal[1] = cum;
}
__global__ void scan_hist_lo(const unsigned* __restrict__ hist, unsigned* scal, int Ksel) {
    unsigned cum = scal[1], selBin = scal[0];
    for (int b = 65535; b >= 0; --b) {
        unsigned c = hist[b];
        if (cum + c >= (unsigned)Ksel) {
            scal[2] = (selBin << 16) | (unsigned)b;
            scal[3] = (unsigned)Ksel - cum;
            return;
        }
        cum += c;
    }
    scal[2] = selBin << 16; scal[3] = 0;
}
// single block, sequential chunked scan: index-ordered ranks for selected set
__global__ __launch_bounds__(1024) void select_topk(const unsigned* __restrict__ keys, int n,
                                                    const unsigned* __restrict__ scal, int* new_id) {
    __shared__ unsigned s_eq[1024];
    __shared__ unsigned s_sel[1024];
    __shared__ unsigned runs[2];
    const unsigned kth = scal[2];
    const int needEq = (int)scal[3];
    const int t = threadIdx.x;
    if (t == 0) { runs[0] = 0; runs[1] = 0; }
    __syncthreads();
    for (int base = 0; base < n; base += 1024) {
        int i = base + t;
        unsigned key = (i < n) ? keys[i] : 0u;
        unsigned eq = (i < n && key == kth) ? 1u : 0u;
        unsigned gt = (i < n && key >  kth) ? 1u : 0u;
        unsigned runEq = runs[0], runSel = runs[1];
        s_eq[t] = eq; __syncthreads();
        for (int off = 1; off < 1024; off <<= 1) {
            unsigned v = (t >= off) ? s_eq[t - off] : 0u; __syncthreads();
            s_eq[t] += v; __syncthreads();
        }
        unsigned eqExcl = s_eq[t] - eq + runEq;
        unsigned sel = (gt || (eq && (int)eqExcl < needEq)) ? 1u : 0u;
        s_sel[t] = sel; __syncthreads();
        for (int off = 1; off < 1024; off <<= 1) {
            unsigned v = (t >= off) ? s_sel[t - off] : 0u; __syncthreads();
            s_sel[t] += v; __syncthreads();
        }
        if (i < n) new_id[i] = sel ? (int)(s_sel[t] - sel + runSel) : -1;
        __syncthreads();
        if (t == 0) { runs[0] = runEq + s_eq[1023]; runs[1] = runSel + s_sel[1023]; }
        __syncthreads();
    }
}
__global__ void gather_scale(const float* __restrict__ x, const float* __restrict__ score,
                             const int* __restrict__ new_id, float* xk, int n) {
    size_t idx = (size_t)blockIdx.x * blockDim.x + threadIdx.x;
    if (idx >= (size_t)n * 128) return;
    int node = (int)(idx >> 7);
    int j = new_id[node];
    if (j < 0) return;
    xk[(size_t)j * 128 + (idx & 127)] = x[idx] * tanhf(score[node]);
}
__global__ void remap_edges(const int* __restrict__ src, const int* __restrict__ dst,
                            const int* __restrict__ new_id, const float* __restrict__ strength,
                            int* s1, int* d1, int* v1, float* strength1, int E) {
    int e = blockIdx.x * blockDim.x + threadIdx.x;
    if (e >= E) return;
    int ns = new_id[src[e]], nd = new_id[dst[e]];
    int v = (ns >= 0 && nd >= 0) ? 1 : 0;
    s1[e] = v ? ns : 0;
    d1[e] = v ? nd : 0;
    v1[e] = v;
    strength1[e] = v ? strength[e] : 0.f;
}
// res = mean over heads; g = res·gate_W + gate_b   (one wave per node)
__global__ __launch_bounds__(256) void head_mean_gate(const float* __restrict__ out1,
                                                      const float* __restrict__ gW,
                                                      const float* __restrict__ gb,
                                                      float* res, float* g, int n) {
    int wv = (blockIdx.x * 256 + threadIdx.x) >> 5;
    int lane = threadIdx.x & 31;
    if (wv >= n) return;
    const float* p = out1 + (size_t)wv * 128;
    float r = 0.25f * (p[lane] + p[32 + lane] + p[64 + lane] + p[96 + lane]);
    res[(size_t)wv * 32 + lane] = r;
    float s = r * gW[lane];
    #pragma unroll
    for (int m = 16; m >= 1; m >>= 1) s += __shfl_xor(s, m, 32);
    if (lane == 0) g[wv] = s + gb[0];
}
// softmax over nodes -> weighted readout -> classifier (single block)
__global__ __launch_bounds__(1024) void readout_cls(const float* __restrict__ res,
                                                    const float* __restrict__ g,
                                                    const float* __restrict__ clsW,
                                                    const float* __restrict__ clsb,
                                                    float* out, int n) {
    __shared__ float red[1024];
    __shared__ float ro[32];
    const int t = threadIdx.x;
    float m = -3.4e38f;
    for (int j = t; j < n; j += 1024) m = fmaxf(m, g[j]);
    red[t] = m; __syncthreads();
    for (int off = 512; off >= 1; off >>= 1) { if (t < off) red[t] = fmaxf(red[t], red[t + off]); __syncthreads(); }
    float gmax = red[0]; __syncthreads();
    float s = 0.f;
    for (int j = t; j < n; j += 1024) s += __expf(g[j] - gmax);
    red[t] = s; __syncthreads();
    for (int off = 512; off >= 1; off >>= 1) { if (t < off) red[t] += red[t + off]; __syncthreads(); }
    float gsum = red[0]; __syncthreads();
    int d = t & 31, rg = t >> 5;
    float acc = 0.f;
    for (int j = rg; j < n; j += 32) acc += (__expf(g[j] - gmax) / gsum) * res[(size_t)j * 32 + d];
    red[t] = acc; __syncthreads();
    if (t < 32) {
        float r = 0.f;
        for (int k = 0; k < 32; ++k) r += red[k * 32 + t];
        out[t] = r;          // readout [1,32]
        ro[t] = r;
    }
    __syncthreads();
    if (t < 2) {
        float l = clsb[t];
        for (int d2 = 0; d2 < 32; ++d2) l += ro[d2] * clsW[d2 * 2 + t];
        out[32 + t] = l;     // lbl_pred [1,2]
    }
}

// ---------------------------------------------------------------------------
extern "C" void kernel_launch(void* const* d_in, const int* in_sizes, int n_in,
                              void* d_out, int out_size, void* d_ws, size_t ws_size,
                              hipStream_t stream) {
    (void)n_in; (void)out_size; (void)ws_size;
    const float* feature  = (const float*)d_in[0];
    const float* strength = (const float*)d_in[1];
    const float* W0   = (const float*)d_in[2];
    const float* b0   = (const float*)d_in[3];
    const float* al0  = (const float*)d_in[4];
    const float* ar0  = (const float*)d_in[5];
    const float* Wc0  = (const float*)d_in[6];
    const float* bc0  = (const float*)d_in[7];
    const float* W1   = (const float*)d_in[8];
    const float* b1   = (const float*)d_in[9];
    const float* al1  = (const float*)d_in[10];
    const float* ar1  = (const float*)d_in[11];
    const float* gW   = (const float*)d_in[12];
    const float* gb   = (const float*)d_in[13];
    const float* cW   = (const float*)d_in[14];
    const float* cb   = (const float*)d_in[15];
    const int*   src  = (const int*)d_in[16];
    const int*   dst  = (const int*)d_in[17];

    const int Nn = in_sizes[0] / 256;  // 50000
    const int Ee = in_sizes[1];        // 800000
    const int Kk = (Nn + 1) / 2;       // 25000

    // output layout: readout[32] | lbl[2] | atten[E*4] | strength1[E]
    float* out      = (float*)d_out;
    float* attenOut = out + 34;
    float* str1Out  = out + 34 + (size_t)Ee * 4;

    // --- workspace bump allocator ---
    char* ws = (char*)d_ws;
    size_t off = 0;
    auto alloc = [&](size_t bytes) -> void* {
        void* p = ws + off;
        off = (off + bytes + 255) & ~(size_t)255;
        return p;
    };
    float*    hfeat0  = (float*)alloc((size_t)Nn * 128 * 4);
    float*    gat0out = (float*)alloc((size_t)Nn * 128 * 4);
    float*    el0     = (float*)alloc((size_t)Nn * 4 * 4);
    float*    er0     = (float*)alloc((size_t)Nn * 4 * 4);
    unsigned* mkey0   = (unsigned*)alloc((size_t)Nn * 4 * 4);
    float*    ssum0   = (float*)alloc((size_t)Nn * 4 * 4);
    float*    ex0     = (float*)alloc((size_t)Ee * 4 * 4);
    float*    dego    = (float*)alloc((size_t)Nn * 4);
    float*    degi    = (float*)alloc((size_t)Nn * 4);
    float*    hc      = (float*)alloc((size_t)Nn * 4);
    float*    agg     = (float*)alloc((size_t)Nn * 4);
    float*    score   = (float*)alloc((size_t)Nn * 4);
    unsigned* keys    = (unsigned*)alloc((size_t)Nn * 4);
    unsigned* hist    = (unsigned*)alloc(65536 * 4);
    unsigned* scal    = (unsigned*)alloc(16 * 4);
    int*      new_id  = (int*)alloc((size_t)Nn * 4);
    float*    xk      = (float*)alloc((size_t)Kk * 128 * 4);
    int*      s1e     = (int*)alloc((size_t)Ee * 4);
    int*      d1e     = (int*)alloc((size_t)Ee * 4);
    int*      v1e     = (int*)alloc((size_t)Ee * 4);
    float*    hfeat1  = (float*)alloc((size_t)Kk * 128 * 4);
    float*    el1     = (float*)alloc((size_t)Kk * 4 * 4);
    float*    er1     = (float*)alloc((size_t)Kk * 4 * 4);
    unsigned* mkey1   = (unsigned*)alloc((size_t)Kk * 4 * 4);
    float*    ssum1   = (float*)alloc((size_t)Kk * 4 * 4);
    float*    ex1     = (float*)alloc((size_t)Ee * 4 * 4);
    float*    out1    = (float*)alloc((size_t)Kk * 128 * 4);
    float*    res     = (float*)alloc((size_t)Kk * 32 * 4);
    float*    gnode   = (float*)alloc((size_t)Kk * 4);

    // host-side fkey(NEG) bit pattern
    float negf = NEGV; unsigned ub; memcpy(&ub, &negf, 4);
    unsigned negkey = ub ^ ((ub >> 31) ? 0xFFFFFFFFu : 0x80000000u);

    auto gridN = [](size_t n, int b) { return (unsigned)((n + b - 1) / b); };

    // ---- GAT layer 0 ----
    gemm_wmma_n128<<<gridN(Nn, 32), 256, 0, stream>>>(feature, W0, hfeat0, Nn, 256);
    attn_proj<<<gridN((size_t)Nn * 4, 256), 256, 0, stream>>>(hfeat0, al0, ar0, el0, er0, Nn);
    fill_u32<<<gridN((size_t)Nn * 4, 256), 256, 0, stream>>>(mkey0, negkey, (size_t)Nn * 4);
    fill_u32<<<gridN((size_t)Nn * 4, 256), 256, 0, stream>>>((unsigned*)ssum0, 0u, (size_t)Nn * 4);
    init_bias<<<gridN((size_t)Nn * 128, 256), 256, 0, stream>>>(gat0out, b0, Nn);
    edge_logit_max<<<gridN((size_t)Ee * 4, 256), 256, 0, stream>>>(src, dst, nullptr, el0, er0, mkey0, Ee);
    edge_exp_sum<<<gridN((size_t)Ee * 4, 256), 256, 0, stream>>>(src, dst, nullptr, el0, er0, mkey0, ex0, ssum0, Ee);
    edge_aggregate<<<gridN(Ee, 2), 256, 0, stream>>>(src, dst, ex0, ssum0, hfeat0, gat0out, nullptr, Ee);

    // ---- SAGPool ----
    fill_u32<<<gridN(Nn, 256), 256, 0, stream>>>((unsigned*)dego, 0u, Nn);
    fill_u32<<<gridN(Nn, 256), 256, 0, stream>>>((unsigned*)degi, 0u, Nn);
    degrees<<<gridN(Ee, 256), 256, 0, stream>>>(src, dst, dego, degi, Ee);
    conv_score_a<<<gridN((size_t)Nn * 32, 256), 256, 0, stream>>>(gat0out, Wc0, dego, hc, Nn);
    fill_u32<<<gridN(Nn, 256), 256, 0, stream>>>((unsigned*)agg, 0u, Nn);
    edge_agg_scalar<<<gridN(Ee, 256), 256, 0, stream>>>(src, dst, hc, agg, Ee);
    score_keys<<<gridN(Nn, 256), 256, 0, stream>>>(agg, degi, bc0, score, keys, Nn);
    fill_u32<<<gridN(65536, 256), 256, 0, stream>>>(hist, 0u, 65536);
    hist_hi<<<gridN(Nn, 256), 256, 0, stream>>>(keys, hist, Nn);
    scan_hist_hi<<<1, 1, 0, stream>>>(hist, scal, Kk);
    fill_u32<<<gridN(65536, 256), 256, 0, stream>>>(hist, 0u, 65536);
    hist_lo<<<gridN(Nn, 256), 256, 0, stream>>>(keys, scal, hist, Nn);
    scan_hist_lo<<<1, 1, 0, stream>>>(hist, scal, Kk);
    select_topk<<<1, 1024, 0, stream>>>(keys, Nn, scal, new_id);
    gather_scale<<<gridN((size_t)Nn * 128, 256), 256, 0, stream>>>(gat0out, score, new_id, xk, Nn);
    remap_edges<<<gridN(Ee, 256), 256, 0, stream>>>(src, dst, new_id, strength, s1e, d1e, v1e, str1Out, Ee);

    // ---- GAT layer 1 (subgraph) ----
    gemm_wmma_n128<<<gridN(Kk, 32), 256, 0, stream>>>(xk, W1, hfeat1, Kk, 128);
    attn_proj<<<gridN((size_t)Kk * 4, 256), 256, 0, stream>>>(hfeat1, al1, ar1, el1, er1, Kk);
    fill_u32<<<gridN((size_t)Kk * 4, 256), 256, 0, stream>>>(mkey1, negkey, (size_t)Kk * 4);
    fill_u32<<<gridN((size_t)Kk * 4, 256), 256, 0, stream>>>((unsigned*)ssum1, 0u, (size_t)Kk * 4);
    init_bias<<<gridN((size_t)Kk * 128, 256), 256, 0, stream>>>(out1, b1, Kk);
    edge_logit_max<<<gridN((size_t)Ee * 4, 256), 256, 0, stream>>>(s1e, d1e, v1e, el1, er1, mkey1, Ee);
    edge_exp_sum<<<gridN((size_t)Ee * 4, 256), 256, 0, stream>>>(s1e, d1e, v1e, el1, er1, mkey1, ex1, ssum1, Ee);
    edge_aggregate<<<gridN(Ee, 2), 256, 0, stream>>>(s1e, d1e, ex1, ssum1, hfeat1, out1, attenOut, Ee);

    // ---- readout + classifier ----
    head_mean_gate<<<gridN((size_t)Kk * 32, 256), 256, 0, stream>>>(out1, gW, gb, res, gnode, Kk);
    readout_cls<<<1, 1024, 0, stream>>>(res, gnode, cW, cb, out, Kk);
}
